// GCN_55147380080825
// MI455X (gfx1250) — matrix operand order repaired and checked by reading, not compiled, not run on hardware
//
#include <hip/hip_runtime.h>
#include <hip/hip_bf16.h>
#include <math.h>

// ---------------------------------------------------------------------------
// Types for CDNA5 WMMA (wave32): A/B = v16bf (8 VGPRs packed bf16),
// C/D = v8f (8 VGPRs f32).
// ---------------------------------------------------------------------------
typedef __attribute__((ext_vector_type(16))) __bf16          v16bf;
typedef __attribute__((ext_vector_type(8)))  float           v8f;
typedef __attribute__((ext_vector_type(8)))  unsigned int    v8u;
typedef __attribute__((ext_vector_type(4)))  float           f4;

__device__ __forceinline__ unsigned short f2bf_rne(float f) {
  unsigned int u = __builtin_bit_cast(unsigned int, f);
  unsigned int r = u + 0x7FFFu + ((u >> 16) & 1u);   // round-to-nearest-even
  return (unsigned short)(r >> 16);
}

// Pack two f32 -> packed 2xbf16 dword: one v_perm_b32 after round-half-up
// biasing (3 VALU per 2 elements instead of ~6 for manual RNE).
__device__ __forceinline__ unsigned int pack2bf(float lo, float hi) {
  unsigned int ul = __builtin_bit_cast(unsigned int, lo) + 0x8000u;
  unsigned int uh = __builtin_bit_cast(unsigned int, hi) + 0x8000u;
  // dst = { uh[31:24], uh[23:16], ul[31:24], ul[23:16] }
  return __builtin_amdgcn_perm(uh, ul, 0x07060302u);
}

// Load one 16x32 A-fragment row-slice for this lane: two contiguous 8-float
// runs (K = ko+half*8..+7 and K = ko+16+half*8..+7) as 4x float4 -> 8 dwords.
__device__ __forceinline__ v16bf load_a_frag(const float* __restrict__ base, bool doRelu) {
  const f4* ap = (const f4*)base;
  f4 q0 = ap[0], q1 = ap[1], q2 = ap[4], q3 = ap[5];
  if (doRelu) {
    #pragma unroll
    for (int i = 0; i < 4; ++i) {
      q0[i] = fmaxf(q0[i], 0.f); q1[i] = fmaxf(q1[i], 0.f);
      q2[i] = fmaxf(q2[i], 0.f); q3[i] = fmaxf(q3[i], 0.f);
    }
  }
  v8u a;
  a[0] = pack2bf(q0[0], q0[1]);  a[1] = pack2bf(q0[2], q0[3]);
  a[2] = pack2bf(q1[0], q1[1]);  a[3] = pack2bf(q1[2], q1[3]);
  a[4] = pack2bf(q2[0], q2[1]);  a[5] = pack2bf(q2[2], q2[3]);
  a[6] = pack2bf(q3[0], q3[1]);  a[7] = pack2bf(q3[2], q3[3]);
  return __builtin_bit_cast(v16bf, a);
}

// ---------------------------------------------------------------------------
// Pack weight matrix B [Ktot x colsValid] (f32, row-major) into fragment-major
// bf16: P[((kchunk*ntiles + ntile)*32 + lane)*16 + elem], zero-padding columns
// up to Nc. After this, each lane's WMMA B-fragment is 32 contiguous bytes.
//   lane = half*16 + (n%16), elem i -> K = kchunk*32 + (i<8 ? half*8 + i : 16 + half*8 + i-8)
// ---------------------------------------------------------------------------
__global__ void pack_b_kernel(const float* __restrict__ B,
                              unsigned short* __restrict__ P,
                              int Ktot, int Nc, int colsValid)
{
  int i = blockIdx.x * blockDim.x + threadIdx.x;
  int total = Ktot * Nc;
  if (i >= total) return;
  int k = i / Nc, n = i % Nc;
  unsigned short v = (n < colsValid) ? f2bf_rne(B[(size_t)k * colsValid + n])
                                     : (unsigned short)0;
  int ntiles = Nc >> 4;
  int kchunk = k >> 5, kin = k & 31;
  int half = (kin >> 3) & 1, elem = (kin & 7) | ((kin >> 4) << 3);
  int ntile = n >> 4, l16 = n & 15;
  int lane = half * 16 + l16;
  P[(((size_t)kchunk * ntiles + ntile) * 32 + lane) * 16 + elem] = v;
}

// ---------------------------------------------------------------------------
// WMMA GEMM:  C[M,Nc] = concat(A0[M,K0T], A1[M,KTOT-K0T]) @ B[KTOT,Nc]
// Compile-time K0T/KTOT/RELU/NT: the K loop fully unrolls so the scheduler
// hoists next-step loads over current-step wmma; A-source select folds away.
// One wave computes a 32(M) x NT*16(N) tile: A loaded+converted ONCE per row,
// 2*NT wmma per K-step.  NT=8 (Nc=128) or NT=4 (Nc=64).
// Block = 256 threads = 8 waves = 8 row-tiles of 32.
// ---------------------------------------------------------------------------
template<int K0T, int KTOT, bool RELU, int NT>
__global__ void __launch_bounds__(256)
gemm_bf16_wmma(const float* __restrict__ A0, const float* __restrict__ A1,
               const unsigned short* __restrict__ Pb, int ntilesB,
               float* __restrict__ C, int ldc, int M)
{
  const int lane = threadIdx.x & 31;
  const int wave = threadIdx.x >> 5;
  const int half = lane >> 4;     // 0: lanes 0-15, 1: lanes 16-31
  const int l16  = lane & 15;

  const int mt = blockIdx.x * 8 + wave;   // 32-row tile index
  const int m0 = mt * 32;
  if (m0 >= M) return;
  const int n0base = blockIdx.y * (16 * NT);

  v8f acc0[NT] = {};
  v8f acc1[NT] = {};

  int row0 = m0 + l16;        if (row0 >= M) row0 = M - 1;
  int row1 = m0 + 16 + l16;   if (row1 >= M) row1 = M - 1;

  const unsigned short* pk0 =
      Pb + ((size_t)(n0base >> 4) * 32 + lane) * 16;

  #pragma unroll
  for (int kc = 0; kc < KTOT; kc += 32) {
    const bool first = (kc < K0T);                      // folds per step
    const float* Asrc = first ? A0 : A1;
    const int lda     = first ? K0T : (KTOT - K0T);
    const int ko      = first ? kc : (kc - K0T);
    const bool doRelu = first && RELU;

    // ---- all loads of this step issued before any wmma
    const unsigned short* pk = pk0 + (size_t)(kc >> 5) * ntilesB * 32 * 16;
    v8u braw[NT];
    #pragma unroll
    for (int t = 0; t < NT; ++t)
      braw[t] = *(const v8u*)(pk + (size_t)t * 32 * 16);  // 32B contiguous

    const int aoff = ko + half * 8;
    v16bf av0 = load_a_frag(Asrc + (size_t)row0 * lda + aoff, doRelu);
    v16bf av1 = load_a_frag(Asrc + (size_t)row1 * lda + aoff, doRelu);

    #pragma unroll
    for (int t = 0; t < NT; ++t) {
      v16bf bv = __builtin_bit_cast(v16bf, braw[t]);
      acc0[t] = __builtin_amdgcn_wmma_f32_16x16x32_bf16(
                    false, av0, false, bv, (short)0, acc0[t], false, false);
      acc1[t] = __builtin_amdgcn_wmma_f32_16x16x32_bf16(
                    false, av1, false, bv, (short)0, acc1[t], false, false);
    }
  }

  // store: C/D layout: VGPR r -> row (half*8 + r) within 16-row subtile
  #pragma unroll
  for (int t = 0; t < NT; ++t) {
    const int colC = n0base + t * 16 + l16;
    #pragma unroll
    for (int r = 0; r < 8; ++r) {
      int rr0 = m0 + half * 8 + r;
      int rr1 = m0 + 16 + half * 8 + r;
      if (rr0 < M) C[(size_t)rr0 * ldc + colC] = acc0[t][r];
      if (rr1 < M) C[(size_t)rr1 * ldc + colC] = acc1[t][r];
    }
  }
}

// ---------------------------------------------------------------------------
// COO SpMM scatter: O[row[e], :] += val[e] * S[col[e], :]
// One wave per edge; float4 gather, f32 atomic scatter into bias-initialized
// accumulator. Working set is L2-resident (192MB L2), so random access is
// L2 traffic rather than HBM.
// ---------------------------------------------------------------------------
__global__ void __launch_bounds__(256)
spmm_scatter(const int* __restrict__ row, const int* __restrict__ col,
             const float* __restrict__ val, const float* __restrict__ S,
             float* __restrict__ O, int E, int F, int ldS, int ldO)
{
  int e = blockIdx.x * (blockDim.x >> 5) + (threadIdx.x >> 5);
  if (e >= E) return;
  int lane = threadIdx.x & 31;
  int r = row[e], c = col[e];
  float v = val[e];
  const f4* sp = (const f4*)(S + (size_t)c * ldS);
  float*   op = O + (size_t)r * ldO;
  int F4 = F >> 2;
  for (int q = lane; q < F4; q += 32) {
    f4 s4 = sp[q];
    atomicAdd(op + q * 4 + 0, v * s4[0]);
    atomicAdd(op + q * 4 + 1, v * s4[1]);
    atomicAdd(op + q * 4 + 2, v * s4[2]);
    atomicAdd(op + q * 4 + 3, v * s4[3]);
  }
}

// O[n*ld + f] = (f < F) ? bias[f] : 0
__global__ void init_bias_kernel(float* __restrict__ O, const float* __restrict__ bias,
                                 size_t total, int F, int ld)
{
  size_t i = (size_t)blockIdx.x * blockDim.x + threadIdx.x;
  if (i >= total) return;
  int f = (int)(i % (size_t)ld);
  O[i] = (f < F) ? bias[f] : 0.0f;
}

// log_softmax over Cc (<=64) classes; one wave per node (wave32 shuffles)
__global__ void __launch_bounds__(256)
logsoftmax_kernel(const float* __restrict__ G, int ld, float* __restrict__ out,
                  int Nn, int Cc)
{
  int n = blockIdx.x * (blockDim.x >> 5) + (threadIdx.x >> 5);
  if (n >= Nn) return;
  int lane = threadIdx.x & 31;
  const float* g = G + (size_t)n * ld;
  float a = (lane      < Cc) ? g[lane]      : -INFINITY;
  float b = (lane + 32 < Cc) ? g[lane + 32] : -INFINITY;
  float m = fmaxf(a, b);
  #pragma unroll
  for (int off = 16; off > 0; off >>= 1)
    m = fmaxf(m, __shfl_xor(m, off, 32));
  float s = ((lane < Cc) ? __expf(a - m) : 0.0f)
          + ((lane + 32 < Cc) ? __expf(b - m) : 0.0f);
  #pragma unroll
  for (int off = 16; off > 0; off >>= 1)
    s += __shfl_xor(s, off, 32);
  float l = m + __logf(s);
  float* o = out + (size_t)n * Cc;
  if (lane      < Cc) o[lane]      = a - l;
  if (lane + 32 < Cc) o[lane + 32] = b - l;
}

// ---------------------------------------------------------------------------
// Launch: 3 reused N*128 f32 buffers keep everything L2-resident.
//   B0 = GEMM out (support), B1 = bias + SpMM accum, B2 = combiner out
// ---------------------------------------------------------------------------
extern "C" void kernel_launch(void* const* d_in, const int* in_sizes, int n_in,
                              void* d_out, int out_size, void* d_ws, size_t ws_size,
                              hipStream_t stream) {
  const float* x   = (const float*)d_in[0];
  const int*   row = (const int*)  d_in[1];
  const int*   col = (const int*)  d_in[2];
  const float* val = (const float*)d_in[3];
  const float* W1  = (const float*)d_in[4];
  const float* b1  = (const float*)d_in[5];
  const float* W2  = (const float*)d_in[6];
  const float* b2  = (const float*)d_in[7];
  const float* W3  = (const float*)d_in[8];
  const float* b3  = (const float*)d_in[9];
  const float* Cmb = (const float*)d_in[10];

  const int F  = 128;                    // F_IN == H
  const int Nn = in_sizes[0] / F;        // 50000
  const int E  = in_sizes[1];            // 800000
  const int Cc = in_sizes[9];            // 40 (nclass)
  const int C3 = 64;                     // padded class width

  float* out = (float*)d_out;

  size_t nb = (size_t)Nn * F;
  float* B0 = (float*)d_ws;
  float* B1 = B0 + nb;
  float* B2 = B1 + nb;
  unsigned short* W1p = (unsigned short*)(B2 + nb);  // packed 128x128
  unsigned short* W2p = W1p + F * F;                 // packed 128x128
  unsigned short* Cp  = W2p + F * F;                 // packed 256x128
  unsigned short* W3p = Cp + 2 * F * F;              // packed 128x64 (pad of 128x40)

  // weight prep: fragment-major bf16 packing (one-time, tiny)
  { int n = F * F;     pack_b_kernel<<<(n + 255) / 256, 256, 0, stream>>>(W1,  W1p, F,     F,  F);  }
  { int n = F * F;     pack_b_kernel<<<(n + 255) / 256, 256, 0, stream>>>(W2,  W2p, F,     F,  F);  }
  { int n = 2 * F * F; pack_b_kernel<<<(n + 255) / 256, 256, 0, stream>>>(Cmb, Cp,  2 * F, F,  F);  }
  { int n = F * C3;    pack_b_kernel<<<(n + 255) / 256, 256, 0, stream>>>(W3,  W3p, F,     C3, Cc); }

  const int mtiles = (Nn + 31) / 32;     // 32-row tiles per wave
  dim3 gGrid((mtiles + 7) / 8);          // full 128-wide strip per wave (NT=8)
  dim3 gGrid3((mtiles + 7) / 8);         // 64-wide strip (NT=4)
  int spmmBlocks = (E + 7) / 8;
  size_t totF = (size_t)Nn * F;
  size_t tot3 = (size_t)Nn * C3;
  int ibF = (int)((totF + 255) / 256);
  int ib3 = (int)((tot3 + 255) / 256);
  const int ntF  = F / 16;               // 8 B-tiles per kchunk (Nc=128)
  const int nt3  = C3 / 16;              // 4 B-tiles per kchunk (Nc=64)

  // ---- layer 1: h = relu(spmm(x@W1) + b1); c1 = [h, x] @ combiner
  gemm_bf16_wmma<128, 128, false, 8><<<gGrid, 256, 0, stream>>>(x, nullptr, W1p, ntF, B0, F, Nn);
  init_bias_kernel<<<ibF, 256, 0, stream>>>(B1, b1, totF, F, F);
  spmm_scatter<<<spmmBlocks, 256, 0, stream>>>(row, col, val, B0, B1, E, F, F, F);
  gemm_bf16_wmma<128, 256, true, 8><<<gGrid, 256, 0, stream>>>(B1, x, Cp, ntF, B2, F, Nn);

  // ---- layer 2: h = spmm(c1@W2) + b2; c2 = [h, x] @ combiner
  gemm_bf16_wmma<128, 128, false, 8><<<gGrid, 256, 0, stream>>>(B2, nullptr, W2p, ntF, B0, F, Nn);
  init_bias_kernel<<<ibF, 256, 0, stream>>>(B1, b2, totF, F, F);
  spmm_scatter<<<spmmBlocks, 256, 0, stream>>>(row, col, val, B0, B1, E, F, F, F);
  gemm_bf16_wmma<128, 256, false, 8><<<gGrid, 256, 0, stream>>>(B1, x, Cp, ntF, B2, F, Nn);

  // ---- layer 3: logits = spmm(c2@W3) + b3; out = log_softmax(logits)
  gemm_bf16_wmma<128, 128, false, 4><<<gGrid3, 256, 0, stream>>>(B2, nullptr, W3p, nt3, B0, C3, Nn);
  init_bias_kernel<<<ib3, 256, 0, stream>>>(B1, b3, tot3, Cc, C3);
  spmm_scatter<<<spmmBlocks, 256, 0, stream>>>(row, col, val, B0, B1, E, Cc, C3, C3);
  logsoftmax_kernel<<<(Nn + 7) / 8, 256, 0, stream>>>(B1, C3, out, Nn, Cc);
}
